// GraphEncoder_40570261078514
// MI455X (gfx1250) — compile-verified, matrix operand order checked by reading, not run on hardware
//
#include <hip/hip_runtime.h>

namespace {
constexpr int N_ = 20000, F_ = 128, H_ = 5, HID_ = 320;
constexpr int NHID_ = 512, NOUT_ = 256, E_ = 320000, G_ = 200;
constexpr int NE_ = E_ + N_;                 // edges including self loops
constexpr unsigned NEG_INF_KEY = 0x007FFFFFu; // monotonic key of -inf
}

typedef __attribute__((ext_vector_type(8)))  float  v8f;
typedef __attribute__((ext_vector_type(2)))  float  v2f;
typedef __attribute__((ext_vector_type(16))) __bf16 v16bf;

#define GEMM_BIAS 1
#define GEMM_RELU 2

#if __has_builtin(__builtin_amdgcn_wmma_f32_16x16x4_f32)
#define USE_F32_WMMA 1
#else
#define USE_F32_WMMA 0
#endif

// ---------------- GEMM: C[M,Nc] = A[M,K] @ W[K,Nc] (+bias)(+relu) ----------
// One wave computes a 32x32 C tile (2x2 WMMA fragments). Nc % 32 == 0,
// K % 32 == 0 (128/320/512). Out-of-range A rows are clamped (their products
// land only in C rows we never store), so loads are unconditional.
__global__ __launch_bounds__(256) void gemm_wmma_kernel(
    const float* __restrict__ A, const float* __restrict__ W,
    const float* __restrict__ bias, float* __restrict__ Cout,
    int M, int K, int Nc, int flags)
{
  const int lane   = threadIdx.x & 31;
  const int wave   = threadIdx.x >> 5;
  const int tilesN = Nc >> 5;
  const int tilesM = (M + 31) >> 5;
  const int tile   = blockIdx.x * 8 + wave;
  if (tile >= tilesM * tilesN) return;     // whole-wave uniform exit
  const int tm   = (tile / tilesN) << 5;
  const int tn   = (tile % tilesN) << 5;
  const int half = lane >> 4;              // lane half selects K group
  const int l15  = lane & 15;
  const int m0   = tm + l15;               // A rows owned by this lane
  const int m1   = tm + 16 + l15;
  const int n0   = tn + l15;               // W columns owned by this lane
  const int mc0  = m0 < M ? m0 : M - 1;    // clamp: garbage rows never stored
  const int mc1  = m1 < M ? m1 : M - 1;
  const float* __restrict__ A0 = A + (size_t)mc0 * K;
  const float* __restrict__ A1 = A + (size_t)mc1 * K;
  v8f acc00 = {}, acc01 = {}, acc10 = {}, acc11 = {};
#if USE_F32_WMMA
  const int kb = half * 2;                 // 16x16x4 f32: lanes16-31 hold K+2,K+3
  for (int k0 = 0; k0 < K; k0 += 4) {
    v2f a0 = *(const v2f*)(A0 + k0 + kb);  // global_load_b64, 8B aligned
    v2f a1 = *(const v2f*)(A1 + k0 + kb);
    const float* Wk = W + (size_t)(k0 + kb) * Nc + n0;
    v2f b0, b1;
    b0.x = Wk[0];  b0.y = Wk[Nc];
    b1.x = Wk[16]; b1.y = Wk[Nc + 16];
    acc00 = __builtin_amdgcn_wmma_f32_16x16x4_f32(false, a0, false, b0, (short)0, acc00, false, false);
    acc01 = __builtin_amdgcn_wmma_f32_16x16x4_f32(false, a0, false, b1, (short)0, acc01, false, false);
    acc10 = __builtin_amdgcn_wmma_f32_16x16x4_f32(false, a1, false, b0, (short)0, acc10, false, false);
    acc11 = __builtin_amdgcn_wmma_f32_16x16x4_f32(false, a1, false, b1, (short)0, acc11, false, false);
  }
#else
  for (int k0 = 0; k0 < K; k0 += 32) {
    v16bf a0, a1, b0, b1;
    const int kbase = k0 + half * 8;       // ISA 16-bit A layout: K split by lane half
#pragma unroll
    for (int e = 0; e < 16; ++e) {
      const int kk = kbase + (e < 8 ? e : 8 + e);
      a0[e] = (__bf16)A0[kk];
      a1[e] = (__bf16)A1[kk];
      const float* Wk = W + (size_t)kk * Nc + n0;
      b0[e] = (__bf16)Wk[0];
      b1[e] = (__bf16)Wk[16];
    }
    acc00 = __builtin_amdgcn_wmma_f32_16x16x32_bf16(false, a0, false, b0, (short)0, acc00, false, false);
    acc01 = __builtin_amdgcn_wmma_f32_16x16x32_bf16(false, a0, false, b1, (short)0, acc01, false, false);
    acc10 = __builtin_amdgcn_wmma_f32_16x16x32_bf16(false, a1, false, b0, (short)0, acc10, false, false);
    acc11 = __builtin_amdgcn_wmma_f32_16x16x32_bf16(false, a1, false, b1, (short)0, acc11, false, false);
  }
#endif
  float bias0 = 0.0f, bias1 = 0.0f;
  if (flags & GEMM_BIAS) { bias0 = bias[n0]; bias1 = bias[n0 + 16]; }
#pragma unroll
  for (int v = 0; v < 8; ++v) {            // C/D: row = v + 8*half, col = l15
    const int r0 = tm + v + half * 8;
    const int r1 = r0 + 16;
    float v00 = acc00[v] + bias0, v01 = acc01[v] + bias1;
    float v10 = acc10[v] + bias0, v11 = acc11[v] + bias1;
    if (flags & GEMM_RELU) {
      v00 = fmaxf(v00, 0.0f); v01 = fmaxf(v01, 0.0f);
      v10 = fmaxf(v10, 0.0f); v11 = fmaxf(v11, 0.0f);
    }
    if (r0 < M) { Cout[(size_t)r0 * Nc + n0] = v00; Cout[(size_t)r0 * Nc + n0 + 16] = v01; }
    if (r1 < M) { Cout[(size_t)r1 * Nc + n0] = v10; Cout[(size_t)r1 * Nc + n0 + 16] = v11; }
  }
}

// ---------------- helpers ----------------
__global__ void fill_u32_kernel(unsigned* __restrict__ p, unsigned v, int n) {
  int i = blockIdx.x * blockDim.x + threadIdx.x;
  if (i < n) p[i] = v;
}

__device__ __forceinline__ float lrelu(float x) { return x > 0.0f ? x : 0.2f * x; }
__device__ __forceinline__ unsigned fkey(float f) {   // order-preserving float->u32
  unsigned u = __float_as_uint(f);
  return (u & 0x80000000u) ? ~u : (u | 0x80000000u);
}
__device__ __forceinline__ float funkey(unsigned k) {
  unsigned u = (k & 0x80000000u) ? (k & 0x7FFFFFFFu) : ~k;
  return __uint_as_float(u);
}
__device__ __forceinline__ void edge_nodes(const int* __restrict__ ei, int e,
                                           int& s, int& d) {
  if (e < E_) { s = ei[e]; d = ei[E_ + e]; } else { s = d = e - E_; }
}

// per-(node,head) attention projections: s = <h, a_src>, d = <h, a_dst>
__global__ void sd_kernel(const float* __restrict__ h, const float* __restrict__ as_,
                          const float* __restrict__ ad_, float* __restrict__ s,
                          float* __restrict__ d) {
  int i = blockIdx.x * blockDim.x + threadIdx.x;
  if (i >= N_ * H_) return;
  int node = i / H_, hd = i - node * H_;
  const float* hp = h + (size_t)node * HID_ + hd * 64;
  const float* a1 = as_ + hd * 64;
  const float* a2 = ad_ + hd * 64;
  float ss = 0.0f, dd = 0.0f;
  for (int c = 0; c < 64; ++c) { float v = hp[c]; ss += v * a1[c]; dd += v * a2[c]; }
  s[i] = ss; d[i] = dd;
}

__global__ void edge_max_kernel(const int* __restrict__ ei, const float* __restrict__ s,
                                const float* __restrict__ d, unsigned* __restrict__ mk) {
  int i = blockIdx.x * blockDim.x + threadIdx.x;
  if (i >= NE_ * H_) return;
  int e = i / H_, hd = i - e * H_;
  int si, di; edge_nodes(ei, e, si, di);
  float sc = lrelu(s[si * H_ + hd] + d[di * H_ + hd]);
  atomicMax(&mk[di * H_ + hd], fkey(sc));
}

__global__ void edge_expsum_kernel(const int* __restrict__ ei, const float* __restrict__ s,
                                   const float* __restrict__ d, const unsigned* __restrict__ mk,
                                   float* __restrict__ den) {
  int i = blockIdx.x * blockDim.x + threadIdx.x;
  if (i >= NE_ * H_) return;
  int e = i / H_, hd = i - e * H_;
  int si, di; edge_nodes(ei, e, si, di);
  int dh = di * H_ + hd;
  float sc = lrelu(s[si * H_ + hd] + d[dh]);
  atomicAdd(&den[dh], expf(sc - funkey(mk[dh])));
}

// per-(edge,head) softmax weight, hoisted out of the wide scatter kernel
__global__ void edge_alpha_kernel(const int* __restrict__ ei, const float* __restrict__ s,
                                  const float* __restrict__ d, const unsigned* __restrict__ mk,
                                  const float* __restrict__ den, float* __restrict__ alpha) {
  int i = blockIdx.x * blockDim.x + threadIdx.x;
  if (i >= NE_ * H_) return;
  int e = i / H_, hd = i - e * H_;
  int si, di; edge_nodes(ei, e, si, di);
  int dh = di * H_ + hd;
  float sc = lrelu(s[si * H_ + hd] + d[dh]);
  alpha[i] = expf(sc - funkey(mk[dh])) / den[dh];
}

// flattened over E'*HID: 64 consecutive threads share one (edge,head) -> coalesced
__global__ void edge_scatter_kernel(const int* __restrict__ ei, const float* __restrict__ alpha,
                                    const float* __restrict__ h, float* __restrict__ agg) {
  long long i = (long long)blockIdx.x * blockDim.x + threadIdx.x;
  if (i >= (long long)NE_ * HID_) return;
  int e  = (int)(i / HID_);
  int hc = (int)(i - (long long)e * HID_);
  int hd = hc >> 6;
  int si, di; edge_nodes(ei, e, si, di);
  float al = alpha[e * H_ + hd];
  atomicAdd(&agg[(size_t)di * HID_ + hc], al * h[(size_t)si * HID_ + hc]);
}

// hres = tmp + relu(agg + b)   (tmp already has its own bias from the GEMM)
__global__ void resid_set_kernel(const float* __restrict__ tmp, const float* __restrict__ agg,
                                 const float* __restrict__ b, float* __restrict__ hres) {
  int i = blockIdx.x * blockDim.x + threadIdx.x;
  if (i >= N_ * HID_) return;
  hres[i] = tmp[i] + fmaxf(agg[i] + b[i % HID_], 0.0f);
}

// hres += relu(agg + b)
__global__ void resid_add_kernel(const float* __restrict__ agg, const float* __restrict__ b,
                                 float* __restrict__ hres) {
  int i = blockIdx.x * blockDim.x + threadIdx.x;
  if (i >= N_ * HID_) return;
  hres[i] += fmaxf(agg[i] + b[i % HID_], 0.0f);
}

__global__ void pool_cnt_kernel(const int* __restrict__ batch, float* __restrict__ cnt) {
  int i = blockIdx.x * blockDim.x + threadIdx.x;
  if (i < N_) atomicAdd(&cnt[batch[i]], 1.0f);
}
__global__ void pool_sum_kernel(const float* __restrict__ hres, const int* __restrict__ batch,
                                float* __restrict__ pooled) {
  int i = blockIdx.x * blockDim.x + threadIdx.x;
  if (i >= N_ * HID_) return;
  int node = i / HID_, f = i - node * HID_;
  atomicAdd(&pooled[(size_t)batch[node] * HID_ + f], hres[i]);
}
__global__ void pool_div_kernel(float* __restrict__ pooled, const float* __restrict__ cnt) {
  int i = blockIdx.x * blockDim.x + threadIdx.x;
  if (i >= G_ * HID_) return;
  pooled[i] /= fmaxf(cnt[i / HID_], 1.0f);
}

// layernorm over NOUT=256 columns; one block per graph row
__global__ __launch_bounds__(NOUT_) void ln_kernel(const float* __restrict__ z,
                                                   const float* __restrict__ g,
                                                   const float* __restrict__ b,
                                                   float* __restrict__ out) {
  __shared__ float red[NOUT_];
  int row = blockIdx.x, t = threadIdx.x;
  float v = z[row * NOUT_ + t];
  red[t] = v; __syncthreads();
  for (int sft = NOUT_ / 2; sft > 0; sft >>= 1) { if (t < sft) red[t] += red[t + sft]; __syncthreads(); }
  float mu = red[0] / (float)NOUT_; __syncthreads();
  float dv = v - mu;
  red[t] = dv * dv; __syncthreads();
  for (int sft = NOUT_ / 2; sft > 0; sft >>= 1) { if (t < sft) red[t] += red[t + sft]; __syncthreads(); }
  float var = red[0] / (float)NOUT_;
  out[row * NOUT_ + t] = g[t] * dv * rsqrtf(var + 1e-5f) + b[t];
}

// ---------------- host orchestration ----------------
extern "C" void kernel_launch(void* const* d_in, const int* in_sizes, int n_in,
                              void* d_out, int out_size, void* d_ws, size_t ws_size,
                              hipStream_t stream) {
  (void)in_sizes; (void)n_in; (void)out_size; (void)ws_size;
  const float* x     = (const float*)d_in[0];
  const int*   ei    = (const int*)d_in[1];   // [2,E]: src then dst
  const int*   batch = (const int*)d_in[2];
  const float *W[5], *as_[5], *ad_[5], *bb[5];
  int idx = 3;
  for (int l = 0; l < 5; ++l) {
    W[l]   = (const float*)d_in[idx++];
    as_[l] = (const float*)d_in[idx++];
    ad_[l] = (const float*)d_in[idx++];
    bb[l]  = (const float*)d_in[idx++];
  }
  const float* Wm1 = (const float*)d_in[idx++]; const float* bm1 = (const float*)d_in[idx++];
  const float* Wm2 = (const float*)d_in[idx++]; const float* bm2 = (const float*)d_in[idx++];
  const float* Wh1 = (const float*)d_in[idx++]; const float* bh1 = (const float*)d_in[idx++];
  const float* Wh2 = (const float*)d_in[idx++]; const float* bh2 = (const float*)d_in[idx++];
  const float* lng = (const float*)d_in[idx++]; const float* lnb = (const float*)d_in[idx++];

  char* ws = (char*)d_ws;
  float*    hbuf   = (float*)ws;    ws += sizeof(float) * (size_t)N_ * HID_;
  float*    agg    = (float*)ws;    ws += sizeof(float) * (size_t)N_ * HID_;
  float*    hres   = (float*)ws;    ws += sizeof(float) * (size_t)N_ * HID_;
  float*    sbuf   = (float*)ws;    ws += sizeof(float) * (size_t)N_ * H_;
  float*    dbuf   = (float*)ws;    ws += sizeof(float) * (size_t)N_ * H_;
  unsigned* mkeys  = (unsigned*)ws; ws += sizeof(unsigned) * (size_t)N_ * H_;
  float*    den    = (float*)ws;    ws += sizeof(float) * (size_t)N_ * H_;
  float*    albuf  = (float*)ws;    ws += sizeof(float) * (size_t)NE_ * H_;
  float*    pooled = (float*)ws;    ws += sizeof(float) * (size_t)G_ * HID_;
  float*    cnt    = (float*)ws;    ws += sizeof(float) * (size_t)G_;
  float*    z1     = (float*)ws;    ws += sizeof(float) * (size_t)G_ * NHID_;
  float*    z2     = (float*)ws;    ws += sizeof(float) * (size_t)G_ * NOUT_;

  auto cdiv = [](long long a, long long b) { return (int)((a + b - 1) / b); };
  auto gemm = [&](const float* A, const float* Wm, const float* bias, float* Cc,
                  int M, int K, int Nc, int flags) {
    int tiles = ((M + 31) / 32) * (Nc / 32);
    gemm_wmma_kernel<<<cdiv(tiles, 8), 256, 0, stream>>>(A, Wm, bias, Cc, M, K, Nc, flags);
  };
  auto gat = [&](const float* Xin, int K, int l) {
    gemm(Xin, W[l], nullptr, hbuf, N_, K, HID_, 0);
    sd_kernel<<<cdiv(N_ * H_, 256), 256, 0, stream>>>(hbuf, as_[l], ad_[l], sbuf, dbuf);
    fill_u32_kernel<<<cdiv(N_ * H_, 256), 256, 0, stream>>>(mkeys, NEG_INF_KEY, N_ * H_);
    fill_u32_kernel<<<cdiv(N_ * H_, 256), 256, 0, stream>>>((unsigned*)den, 0u, N_ * H_);
    fill_u32_kernel<<<cdiv((long long)N_ * HID_, 256), 256, 0, stream>>>((unsigned*)agg, 0u, N_ * HID_);
    edge_max_kernel<<<cdiv((long long)NE_ * H_, 256), 256, 0, stream>>>(ei, sbuf, dbuf, mkeys);
    edge_expsum_kernel<<<cdiv((long long)NE_ * H_, 256), 256, 0, stream>>>(ei, sbuf, dbuf, mkeys, den);
    edge_alpha_kernel<<<cdiv((long long)NE_ * H_, 256), 256, 0, stream>>>(ei, sbuf, dbuf, mkeys, den, albuf);
    edge_scatter_kernel<<<cdiv((long long)NE_ * HID_, 256), 256, 0, stream>>>(ei, albuf, hbuf, agg);
  };

  // layer 1: x0 = relu(gat(x)); h = (x@Wm1+bm1) + x0
  gat(x, F_, 0);
  gemm(x, Wm1, bm1, hbuf, N_, F_, HID_, GEMM_BIAS);   // hbuf free after scatter
  resid_set_kernel<<<cdiv(N_ * HID_, 256), 256, 0, stream>>>(hbuf, agg, bb[0], hres);

  // layer 2: x1 = relu(gat(h)); h = (h@Wm2+bm2) + x1
  gat(hres, HID_, 1);
  gemm(hres, Wm2, bm2, hbuf, N_, HID_, HID_, GEMM_BIAS);
  resid_set_kernel<<<cdiv(N_ * HID_, 256), 256, 0, stream>>>(hbuf, agg, bb[1], hres);

  // layers 3-5: h = h + relu(gat(h))
  for (int l = 2; l < 5; ++l) {
    gat(hres, HID_, l);
    resid_add_kernel<<<cdiv(N_ * HID_, 256), 256, 0, stream>>>(agg, bb[l], hres);
  }

  // global mean pool
  fill_u32_kernel<<<cdiv(G_, 256), 256, 0, stream>>>((unsigned*)cnt, 0u, G_);
  fill_u32_kernel<<<cdiv(G_ * HID_, 256), 256, 0, stream>>>((unsigned*)pooled, 0u, G_ * HID_);
  pool_cnt_kernel<<<cdiv(N_, 256), 256, 0, stream>>>(batch, cnt);
  pool_sum_kernel<<<cdiv((long long)N_ * HID_, 256), 256, 0, stream>>>(hres, batch, pooled);
  pool_div_kernel<<<cdiv(G_ * HID_, 256), 256, 0, stream>>>(pooled, cnt);

  // head MLP + layernorm
  gemm(pooled, Wh1, bh1, z1, G_, HID_, NHID_, GEMM_BIAS | GEMM_RELU);
  gemm(z1, Wh2, bh2, z2, G_, NHID_, NOUT_, GEMM_BIAS);
  ln_kernel<<<G_, NOUT_, 0, stream>>>(z2, lng, lnb, (float*)d_out);
}
